// MultiHeadAttention_53618371724024
// MI455X (gfx1250) — matrix operand order repaired and checked by reading, not compiled, not run on hardware
//
#include <hip/hip_runtime.h>
#include <hip/hip_bf16.h>

// ---------------------------------------------------------------------------
// MHA forward for MI455X (gfx1250): bf16 WMMA everywhere, f32 accumulate.
// B=2, T=2048, D=1024, H=16, HD=64
// Stage 1/3: double-buffered LDS staging of weight tiles via
// global_load_async_to_lds_b128 (ASYNCcnt); B tiles consumed through a 2-deep
// register pipeline; amdgpu_waves_per_eu(1) frees VGPRs so the pipeline's
// ping-pong tiles get distinct registers (partial dscnt waits).
// ---------------------------------------------------------------------------

#define BB 2
#define TT 2048
#define DD 1024
#define HH 16
#define HDIM 64

typedef __bf16 bf16;
typedef __attribute__((ext_vector_type(16))) __bf16 v16bf;
typedef __attribute__((ext_vector_type(8)))  __bf16 bf16x8;
typedef __attribute__((ext_vector_type(8)))  float  v8f;
typedef __attribute__((ext_vector_type(4)))  int    v4i;

union V16U { v16bf v; bf16x8 h[2]; };

#if __has_builtin(__builtin_amdgcn_global_load_async_to_lds_b128)
#define HAS_ASYNC_LDS 1
typedef __attribute__((address_space(1))) v4i as1_v4i;
typedef __attribute__((address_space(3))) v4i as3_v4i;
#else
#define HAS_ASYNC_LDS 0
#endif

// 16-byte global -> LDS copy. Async (ASYNCcnt) when available, else via VGPRs.
__device__ __forceinline__ void copy_b128_to_lds(const bf16* g, bf16* l) {
#if HAS_ASYNC_LDS
    __builtin_amdgcn_global_load_async_to_lds_b128(
        (as1_v4i*)(uintptr_t)g,
        (as3_v4i*)(uint32_t)(uintptr_t)l,   // low 32 bits of generic == LDS offset
        0, 0);
#else
    *(bf16x8*)l = *(const bf16x8*)g;
#endif
}

__device__ __forceinline__ void lds_wait_async() {
#if HAS_ASYNC_LDS
#if __has_builtin(__builtin_amdgcn_s_wait_asynccnt)
    __builtin_amdgcn_s_wait_asynccnt(0);
#else
    asm volatile("s_wait_asynccnt 0x0" ::: "memory");
#endif
#endif
}

// --- A-operand tile load: 16x32 bf16, row-major source, row stride ld.
// ISA layout: lane L -> M = L%16 ; K set = (L<16) ? {0..7,16..23} : {8..15,24..31}
__device__ __forceinline__ v16bf load_tile_a(const bf16* __restrict__ p, int ld) {
    const int lane = threadIdx.x & 31;
    const bf16* r = p + (size_t)(lane & 15) * ld + (lane >> 4) * 8;
    V16U t;
    t.h[0] = *(const bf16x8*)(r);
    t.h[1] = *(const bf16x8*)(r + 16);
    return t.v;
}

// --- B-operand tile load from B^T stored row-major (N rows x K cols), stride ld.
// Layout: lane L -> N = L%16 ; K set = (L<16) ? {0..7,16..23} : {8..15,24..31}
__device__ __forceinline__ v16bf load_tile_b(const bf16* __restrict__ p, int ld) {
    const int lane = threadIdx.x & 31;
    const bf16* r = p + (size_t)(lane & 15) * ld + (lane >> 4) * 8;
    V16U t;
    t.h[0] = *(const bf16x8*)(r);
    t.h[1] = *(const bf16x8*)(r + 16);
    return t.v;
}

// --- B tile from an LDS-resident 16x32 row-major tile (32-elem row stride).
__device__ __forceinline__ v16bf load_tile_b_lds(const bf16* p) {
    const int lane = threadIdx.x & 31;
    const bf16* r = p + (lane & 15) * 32 + (lane >> 4) * 8;
    V16U t;
    t.h[0] = *(const bf16x8*)(r);
    t.h[1] = *(const bf16x8*)(r + 16);
    return t.v;
}

__device__ __forceinline__ v8f wmma_bf16(v16bf a, v16bf b, v8f c) {
    return __builtin_amdgcn_wmma_f32_16x16x32_bf16(
        false, a, false, b, (short)0, c, false, false);
}

// ---------------------------------------------------------------------------
// Stage 0: dtype conversion kernels
// ---------------------------------------------------------------------------
__global__ void cvt_kernel(const float* __restrict__ in, bf16* __restrict__ out, int n) {
    int i = blockIdx.x * blockDim.x + threadIdx.x;
    if (i < n) out[i] = (bf16)in[i];
}

// W [H, D, HD]  ->  WT [H, HD, D]  (output-indexed for coalesced stores)
__global__ void cvtT_kernel(const float* __restrict__ w, bf16* __restrict__ wT) {
    int j = blockIdx.x * blockDim.x + threadIdx.x;   // over H*HD*D
    int h   = j >> 16;
    int rem = j & 65535;
    int n   = rem >> 10;
    int kd  = rem & 1023;
    wT[j] = (bf16)w[((size_t)h * DD + kd) * HDIM + n];
}

// ---------------------------------------------------------------------------
// Stage 1: QKV projection.  grid = (T/128, B*H), block = 256 (8 waves).
// Weight tiles double-buffered through LDS (12 KB/step), K loop unrolled x2.
// acc order: [c][q,k,v] -> acc[3*c + w].
// ---------------------------------------------------------------------------
__attribute__((amdgpu_waves_per_eu(1)))
__global__ void qkv_kernel(const bf16* __restrict__ x,    // [B,T,D]
                           const bf16* __restrict__ wqT,  // [H,HD,D]
                           const bf16* __restrict__ wkT,
                           const bf16* __restrict__ wvT,
                           bf16* __restrict__ q,          // [B,H,T,HD]
                           bf16* __restrict__ k,
                           bf16* __restrict__ vT)         // [B,H,HD,T]
{
    __shared__ __align__(16) bf16 wlds[2][3 * 4 * 512];   // 2 x 12 KB

    const int bh   = blockIdx.y;
    const int b    = bh >> 4, h = bh & 15;
    const int tid  = threadIdx.x;
    const int wave = tid >> 5, lane = tid & 31;
    const int hi   = lane >> 4, col = lane & 15;
    const int trow = blockIdx.x * 128 + wave * 16;

    const bf16*  xb   = x + ((size_t)b * TT + trow) * DD;
    const size_t hoff = (size_t)h * HDIM * DD;

    // staging map: thread -> (n row 0..63, 8-elem k sub-chunk 0..3)
    const int sn  = tid >> 2;
    const int skc = tid & 3;
    const int doff = (sn >> 4) * 512 + (sn & 15) * 32 + skc * 8;
    const size_t soff = (size_t)sn * DD + skc * 8;
    const bf16* gq = wqT + hoff + soff;
    const bf16* gk = wkT + hoff + soff;
    const bf16* gv = wvT + hoff + soff;

    v8f acc[12] = {};

    auto stage = [&](int buf, int k0) {
        copy_b128_to_lds(gq + k0, &wlds[buf][0 * 2048 + doff]);
        copy_b128_to_lds(gk + k0, &wlds[buf][1 * 2048 + doff]);
        copy_b128_to_lds(gv + k0, &wlds[buf][2 * 2048 + doff]);
    };

    // tile i = (c = i/3, w = i%3) at LDS offset w*2048 + c*512
    constexpr int toff[12] = {    0, 2048, 4096,
                                512, 2560, 4608,
                               1024, 3072, 5120,
                               1536, 3584, 5632 };

    // 2-deep pipelined B-tile consumption: before WMMA i, issue loads of
    // tile i+1 -> each WMMA needs only s_wait_dscnt <= 2.
    auto compute = [&](const bf16* lb, v16bf a) {
        v16bf tcur = load_tile_b_lds(lb + toff[0]);
#pragma unroll
        for (int i = 0; i < 12; ++i) {
            v16bf tnext = tcur;
            if (i + 1 < 12) tnext = load_tile_b_lds(lb + toff[i + 1]);
            acc[i] = wmma_bf16(a, tcur, acc[i]);
            tcur = tnext;
        }
    };

    const int nsteps = DD / 32;                      // 32 (even)
    stage(0, 0);
    v16bf aA = load_tile_a(xb, DD);

    for (int it = 0; it < nsteps; it += 2) {
        // ---- even step: consume buf 0, stage buf 1
        lds_wait_async();
        __syncthreads();
        stage(1, (it + 1) * 32);
        v16bf aB = load_tile_a(xb + (it + 1) * 32, DD);
        compute(&wlds[0][0], aA);

        // ---- odd step: consume buf 1, stage buf 0
        lds_wait_async();
        __syncthreads();
        if (it + 2 < nsteps) {
            stage(0, (it + 2) * 32);
            aA = load_tile_a(xb + (it + 2) * 32, DD);
        }
        compute(&wlds[1][0], aB);
    }

    const size_t base = ((size_t)b * HH + h);
#pragma unroll
    for (int c = 0; c < 4; ++c) {
#pragma unroll
        for (int r = 0; r < 8; ++r) {
            const int tr = trow + r + 8 * hi;
            const int n  = c * 16 + col;
            q[(base * TT + tr) * HDIM + n]  = (bf16)acc[3 * c + 0][r];
            k[(base * TT + tr) * HDIM + n]  = (bf16)acc[3 * c + 1][r];
            vT[(base * HDIM + n) * TT + tr] = (bf16)acc[3 * c + 2][r];
        }
    }
}

// ---------------------------------------------------------------------------
// Stage 2: causal flash attention.  grid = (T/128, B*H), block = 256.
// Each wave: 16 query rows; online softmax over key blocks of 32.
// ---------------------------------------------------------------------------
__global__ void attn_kernel(const bf16* __restrict__ q,   // [B,H,T,HD]
                            const bf16* __restrict__ kk,  // [B,H,T,HD]
                            const bf16* __restrict__ vT,  // [B,H,HD,T]
                            bf16* __restrict__ concat)    // [B,T,H*HD]
{
    __shared__ __align__(16) bf16 pbuf[8][16 * 32];

    const int bh   = blockIdx.y;
    const int b    = bh >> 4, h = bh & 15;
    const int wave = threadIdx.x >> 5;
    const int lane = threadIdx.x & 31;
    const int hi   = lane >> 4;
    const int col  = lane & 15;
    const int q0   = blockIdx.x * 128 + wave * 16;

    const bf16* qp = q  + (((size_t)b * HH + h) * TT + q0) * HDIM;
    const bf16* kp = kk + ((size_t)b * HH + h) * TT * HDIM;
    const bf16* vp = vT + ((size_t)b * HH + h) * HDIM * TT;

    const v16bf a0 = load_tile_a(qp,      HDIM);  // K = 0..31
    const v16bf a1 = load_tile_a(qp + 32, HDIM);  // K = 32..63

    v8f o[4] = {};
    float m[8], l[8];
#pragma unroll
    for (int r = 0; r < 8; ++r) { m[r] = -1e30f; l[r] = 0.0f; }

    const float scale = 0.125f;  // HD^-0.5

    for (int j = 0; j <= q0 + 15; j += 32) {
        // ---- S = Q * K^T for 32 keys (two 16x16 tiles); B^T == K rows.
        v8f s0 = {}, s1 = {};
        {
            v16bf b00 = load_tile_b(kp + (size_t)j * HDIM,             HDIM);
            v16bf b01 = load_tile_b(kp + (size_t)j * HDIM + 32,        HDIM);
            s0 = wmma_bf16(a0, b00, s0);
            s0 = wmma_bf16(a1, b01, s0);
            v16bf b10 = load_tile_b(kp + (size_t)(j + 16) * HDIM,      HDIM);
            v16bf b11 = load_tile_b(kp + (size_t)(j + 16) * HDIM + 32, HDIM);
            s1 = wmma_bf16(a0, b10, s1);
            s1 = wmma_bf16(a1, b11, s1);
        }

        // ---- issue V^T tile loads early; latency hides under softmax VALU
        v16bf bv0 = load_tile_b(vp + (size_t)(0 * 16) * TT + j, TT);
        v16bf bv1 = load_tile_b(vp + (size_t)(1 * 16) * TT + j, TT);
        v16bf bv2 = load_tile_b(vp + (size_t)(2 * 16) * TT + j, TT);
        v16bf bv3 = load_tile_b(vp + (size_t)(3 * 16) * TT + j, TT);

        // ---- mask + scale + online softmax stats
        float alpha[8];
#pragma unroll
        for (int r = 0; r < 8; ++r) {
            const int qi = q0 + r + 8 * hi;
            float x0 = (j + col      <= qi) ? s0[r] * scale : -1e30f;
            float x1 = (j + 16 + col <= qi) ? s1[r] * scale : -1e30f;
            float mx = fmaxf(x0, x1);
#pragma unroll
            for (int msk = 1; msk < 16; msk <<= 1)
                mx = fmaxf(mx, __shfl_xor(mx, msk, 32));
            const float mn = fmaxf(m[r], mx);
            const float al = __expf(m[r] - mn);
            const float p0 = __expf(x0 - mn);
            const float p1 = __expf(x1 - mn);
            float ps = p0 + p1;
#pragma unroll
            for (int msk = 1; msk < 16; msk <<= 1)
                ps += __shfl_xor(ps, msk, 32);
            l[r] = l[r] * al + ps;
            m[r] = mn;
            alpha[r] = al;
            s0[r] = p0;
            s1[r] = p1;
        }

        // ---- rescale running O
#pragma unroll
        for (int c = 0; c < 4; ++c)
#pragma unroll
            for (int r = 0; r < 8; ++r)
                o[c][r] *= alpha[r];

        // ---- P (C-layout f32) -> LDS bf16 16x32 -> A-layout regs
        bf16* pw = &pbuf[wave][0];
#pragma unroll
        for (int r = 0; r < 8; ++r) {
            const int row = r + 8 * hi;
            pw[row * 32 + col]      = (bf16)s0[r];
            pw[row * 32 + 16 + col] = (bf16)s1[r];
        }
        __builtin_amdgcn_wave_barrier();  // same-wave LDS ops in-order
        v16bf pa;
        {
            const bf16* rr = pw + (lane & 15) * 32 + hi * 8;
            V16U t;
            t.h[0] = *(const bf16x8*)(rr);
            t.h[1] = *(const bf16x8*)(rr + 16);
            pa = t.v;
        }

        // ---- O += P (16x32) * V (32x64)
        o[0] = wmma_bf16(pa, bv0, o[0]);
        o[1] = wmma_bf16(pa, bv1, o[1]);
        o[2] = wmma_bf16(pa, bv2, o[2]);
        o[3] = wmma_bf16(pa, bv3, o[3]);
    }

    // ---- normalize and store to concat [B,T,H*HD]
#pragma unroll
    for (int c = 0; c < 4; ++c) {
#pragma unroll
        for (int r = 0; r < 8; ++r) {
            const int tr = q0 + r + 8 * hi;
            const int n  = h * HDIM + c * 16 + col;
            concat[((size_t)b * TT + tr) * (HH * HDIM) + n] = (bf16)(o[c][r] / l[r]);
        }
    }
}

// ---------------------------------------------------------------------------
// Stage 3: output projection.  y = concat @ Wp.T + bp
// B^T (NxK) == Wp rows; Wp tiles double-buffered through LDS (4 KB/step).
// grid = (B*T/128, D/64), block 256.
// ---------------------------------------------------------------------------
__attribute__((amdgpu_waves_per_eu(1)))
__global__ void proj_kernel(const bf16* __restrict__ concat, // [B*T, 1024]
                            const bf16* __restrict__ wp,     // [1024, 1024] bf16
                            const float* __restrict__ bp,    // [1024]
                            float* __restrict__ y)           // [B*T, 1024]
{
    __shared__ __align__(16) bf16 plds[2][4 * 512];          // 2 x 4 KB

    const int tid   = threadIdx.x;
    const int wave  = tid >> 5, lane = tid & 31;
    const int hi    = lane >> 4, col = lane & 15;
    const int mrow  = blockIdx.x * 128 + wave * 16;
    const int ncol0 = blockIdx.y * 64;

    const bf16* ap = concat + (size_t)mrow * (HH * HDIM);

    const int sn  = tid >> 2;
    const int skc = tid & 3;
    const int doff = (sn >> 4) * 512 + (sn & 15) * 32 + skc * 8;
    const bf16* gw = wp + (size_t)(ncol0 + sn) * (HH * HDIM) + skc * 8;

    v8f acc[4] = {};

    auto stage = [&](int buf, int k0) {
        copy_b128_to_lds(gw + k0, &plds[buf][doff]);
    };

    // 2-deep pipelined B-tile consumption.
    auto compute = [&](const bf16* lb, v16bf a) {
        v16bf tcur = load_tile_b_lds(lb);
#pragma unroll
        for (int i = 0; i < 4; ++i) {
            v16bf tnext = tcur;
            if (i + 1 < 4) tnext = load_tile_b_lds(lb + (i + 1) * 512);
            acc[i] = wmma_bf16(a, tcur, acc[i]);
            tcur = tnext;
        }
    };

    const int nsteps = DD / 32;                      // 32 (even)
    stage(0, 0);
    v16bf aA = load_tile_a(ap, HH * HDIM);

    for (int it = 0; it < nsteps; it += 2) {
        lds_wait_async();
        __syncthreads();
        stage(1, (it + 1) * 32);
        v16bf aB = load_tile_a(ap + (it + 1) * 32, HH * HDIM);
        compute(&plds[0][0], aA);

        lds_wait_async();
        __syncthreads();
        if (it + 2 < nsteps) {
            stage(0, (it + 2) * 32);
            aA = load_tile_a(ap + (it + 2) * 32, HH * HDIM);
        }
        compute(&plds[1][0], aB);
    }

#pragma unroll
    for (int c = 0; c < 4; ++c) {
#pragma unroll
        for (int r = 0; r < 8; ++r) {
            const int mm = mrow + r + 8 * hi;
            const int nn = ncol0 + c * 16 + col;
            y[(size_t)mm * DD + nn] = acc[c][r] + bp[nn];
        }
    }
}

// ---------------------------------------------------------------------------
extern "C" void kernel_launch(void* const* d_in, const int* in_sizes, int n_in,
                              void* d_out, int out_size, void* d_ws, size_t ws_size,
                              hipStream_t stream) {
    const float* x  = (const float*)d_in[0];
    const float* Wq = (const float*)d_in[1];
    const float* Wk = (const float*)d_in[2];
    const float* Wv = (const float*)d_in[3];
    const float* Wp = (const float*)d_in[4];
    const float* bp = (const float*)d_in[5];
    float* out = (float*)d_out;

    bf16* ws = (bf16*)d_ws;
    size_t o = 0;
    bf16* xb  = ws + o; o += (size_t)BB * TT * DD;        // 4M
    bf16* wqT = ws + o; o += (size_t)HH * HDIM * DD;      // 1M
    bf16* wkT = ws + o; o += (size_t)HH * HDIM * DD;
    bf16* wvT = ws + o; o += (size_t)HH * HDIM * DD;
    bf16* wpb = ws + o; o += (size_t)DD * HH * HDIM;      // 1M
    bf16* qb  = ws + o; o += (size_t)BB * HH * TT * HDIM; // 4M
    bf16* kb  = ws + o; o += (size_t)BB * HH * TT * HDIM;
    bf16* vTb = ws + o; o += (size_t)BB * HH * HDIM * TT;
    bf16* cat = ws + o; o += (size_t)BB * TT * HH * HDIM;
    (void)ws_size; (void)in_sizes; (void)n_in; (void)out_size;

    const int nx = BB * TT * DD;          // 4,194,304
    const int nw = HH * DD * HDIM;        // 1,048,576

    cvt_kernel <<<dim3((nx + 255) / 256), 256, 0, stream>>>(x,  xb,  nx);
    cvtT_kernel<<<dim3(nw / 256),         256, 0, stream>>>(Wq, wqT);
    cvtT_kernel<<<dim3(nw / 256),         256, 0, stream>>>(Wk, wkT);
    cvtT_kernel<<<dim3(nw / 256),         256, 0, stream>>>(Wv, wvT);
    cvt_kernel <<<dim3(nw / 256),         256, 0, stream>>>(Wp, wpb, nw);

    qkv_kernel <<<dim3(TT / 128, BB * HH), 256, 0, stream>>>(xb, wqT, wkT, wvT,
                                                             qb, kb, vTb);
    attn_kernel<<<dim3(TT / 128, BB * HH), 256, 0, stream>>>(qb, kb, vTb, cat);
    proj_kernel<<<dim3((BB * TT) / 128, DD / 64), 256, 0, stream>>>(cat, wpb, bp, out);
}
